// MultiHeadAttention_1365799600649
// MI455X (gfx1250) — compile-verified, matrix operand order checked by reading, not run on hardware
//
#include <hip/hip_runtime.h>
#include <hip/hip_fp16.h>

typedef __attribute__((ext_vector_type(16))) _Float16 v16h;
typedef __attribute__((ext_vector_type(8)))  _Float16 v8h;
typedef __attribute__((ext_vector_type(4)))  _Float16 v4h;
typedef __attribute__((ext_vector_type(8)))  float    v8f;
typedef __attribute__((ext_vector_type(4)))  float    v4f;

#define HIDDEN  768
#define NHEADS  12
#define HEADD   64
#define SEQ     4096
#define BATCH   2
#define ROWS    (BATCH*SEQ)     /* 8192 */
#define QKV_OUT (3*HIDDEN)      /* 2304 */

// ---- Fragment loader (CDNA5 16-bit A/B layout for V_WMMA_F32_16X16X32_F16) ----
// Per lane: row = lane&15, K-half = lane>>4. Elements 0..7 = 8 contiguous halfs
// at row[kh*8]; elements 8..15 = 8 contiguous halfs at row[16+kh*8]
// -> exactly two global_load_b128 per fragment, no VALU.
__device__ inline v16h frag_ld_f16(const _Float16* p, int ld, int lane) {
  const _Float16* row = p + (size_t)(lane & 15) * ld + ((lane >> 4) & 1) * 8;
  const v8h lo = *(const v8h*)(row);
  const v8h hi = *(const v8h*)(row + 16);
  return __builtin_shufflevector(lo, hi, 0, 1, 2, 3, 4, 5, 6, 7,
                                         8, 9, 10, 11, 12, 13, 14, 15);
}

__device__ inline v8f wmma_f16(v16h a, v16h b, v8f c) {
  return __builtin_amdgcn_wmma_f32_16x16x32_f16(
      /*neg_a=*/false, a, /*neg_b=*/false, b,
      /*c_mod=*/(short)0, c, /*reuse_a=*/false, /*reuse_b=*/false);
}

// ------------- Kernel 0: one-time fp32 -> f16 conversion -------------
// Amortizes conversion to O(elements) instead of O(tiles * K) in the GEMMs.
__global__ __launch_bounds__(256)
void cvt_kernel(const float* __restrict__ src, _Float16* __restrict__ dst, int n4) {
  const int i = (blockIdx.x * 256 + threadIdx.x);
  if (i < n4) {
    const v4f s = *(const v4f*)(src + (size_t)i * 4);
    v4h d;
    d[0] = (_Float16)s[0];   // packs to v_cvt_pk_f16_f32
    d[1] = (_Float16)s[1];
    d[2] = (_Float16)s[2];
    d[3] = (_Float16)s[3];
    *(v4h*)(dst + (size_t)i * 4) = d;
  }
}

// ------------- Kernel 1: QKV projection, 32x64 tile per wave -------------
// 12 b128 loads -> 8 WMMAs per k-step. One 64-wide tile = one head of Q/K/V,
// so all output routing is wave-uniform scalar math.
__global__ __launch_bounds__(256)
void qkv_kernel(const _Float16* __restrict__ x16, const _Float16* __restrict__ w16,
                _Float16* __restrict__ qbuf, _Float16* __restrict__ kbuf,
                _Float16* __restrict__ vtbuf) {
  const int lane   = threadIdx.x & 31;
  const int wid    = blockIdx.x * 8 + (threadIdx.x >> 5);
  const int tile_m = wid % (ROWS / 32);      // 0..255
  const int tn64   = wid / (ROWS / 32);      // 0..35
  const _Float16* arow = x16 + (size_t)tile_m * 32 * HIDDEN;
  const _Float16* brow = w16 + (size_t)tn64 * 64 * HIDDEN;

  v8f acc[2][4] = {};
  for (int k0 = 0; k0 < HIDDEN; k0 += 32) {
    __builtin_prefetch(arow + k0 + 64, 0, 1);   // global_prefetch_b8
    v16h a0 = frag_ld_f16(arow + k0,               HIDDEN, lane);
    v16h a1 = frag_ld_f16(arow + 16 * HIDDEN + k0, HIDDEN, lane);
    v16h b0 = frag_ld_f16(brow + k0,               HIDDEN, lane);
    v16h b1 = frag_ld_f16(brow + 16 * HIDDEN + k0, HIDDEN, lane);
    v16h b2 = frag_ld_f16(brow + 32 * HIDDEN + k0, HIDDEN, lane);
    v16h b3 = frag_ld_f16(brow + 48 * HIDDEN + k0, HIDDEN, lane);
    acc[0][0] = wmma_f16(a0, b0, acc[0][0]);
    acc[0][1] = wmma_f16(a0, b1, acc[0][1]);
    acc[0][2] = wmma_f16(a0, b2, acc[0][2]);
    acc[0][3] = wmma_f16(a0, b3, acc[0][3]);
    acc[1][0] = wmma_f16(a1, b0, acc[1][0]);
    acc[1][1] = wmma_f16(a1, b1, acc[1][1]);
    acc[1][2] = wmma_f16(a1, b2, acc[1][2]);
    acc[1][3] = wmma_f16(a1, b3, acc[1][3]);
  }

  const int kh    = (lane >> 4) & 1;
  const int cl    = lane & 15;
  const int which = tn64 / NHEADS;              // 0=Q 1=K 2=V (scalar)
  const int h     = tn64 % NHEADS;              // head (scalar)
  const int bb    = tile_m / (SEQ / 32);        // batch (scalar)
  const int nbase = (tile_m % (SEQ / 32)) * 32; // seq base (scalar)
  const size_t bh = (size_t)(bb * NHEADS + h);

  #pragma unroll
  for (int mi = 0; mi < 2; ++mi) {
    #pragma unroll
    for (int v = 0; v < 8; ++v) {
      const int n = nbase + mi * 16 + v + 8 * kh;
      if (which == 0) {
        _Float16* q = qbuf + (bh * SEQ + n) * HEADD;
        q[ 0 + cl] = (_Float16)acc[mi][0][v];
        q[16 + cl] = (_Float16)acc[mi][1][v];
        q[32 + cl] = (_Float16)acc[mi][2][v];
        q[48 + cl] = (_Float16)acc[mi][3][v];
      } else if (which == 1) {
        _Float16* k = kbuf + (bh * SEQ + n) * HEADD;
        k[ 0 + cl] = (_Float16)acc[mi][0][v];
        k[16 + cl] = (_Float16)acc[mi][1][v];
        k[32 + cl] = (_Float16)acc[mi][2][v];
        k[48 + cl] = (_Float16)acc[mi][3][v];
      } else {  // V stored transposed: [b,h,d,n]
        _Float16* vt = vtbuf + bh * HEADD * SEQ + n;
        vt[(size_t)( 0 + cl) * SEQ] = (_Float16)acc[mi][0][v];
        vt[(size_t)(16 + cl) * SEQ] = (_Float16)acc[mi][1][v];
        vt[(size_t)(32 + cl) * SEQ] = (_Float16)acc[mi][2][v];
        vt[(size_t)(48 + cl) * SEQ] = (_Float16)acc[mi][3][v];
      }
    }
  }
}

// ----------- Kernel 2: flash attention, one wave per 16-row Q tile -----------
__global__ __launch_bounds__(256)
void attn_kernel(const _Float16* __restrict__ qbuf, const _Float16* __restrict__ kbuf,
                 const _Float16* __restrict__ vtbuf, _Float16* __restrict__ attnbuf) {
  __shared__ __align__(16) _Float16 lds_p[8][16 * 32];  // per-wave P tile
  const int lane  = threadIdx.x & 31;
  const int wslot = threadIdx.x >> 5;
  const int wid   = blockIdx.x * 8 + wslot;
  const int mt    = wid % (SEQ / 16);
  const int bh    = wid / (SEQ / 16);      // 0..23
  const int n0    = mt * 16;

  const _Float16* qtile  = qbuf  + ((size_t)bh * SEQ + n0) * HEADD;
  const _Float16* kbase  = kbuf  + (size_t)bh * SEQ * HEADD;
  const _Float16* vtbase = vtbuf + (size_t)bh * HEADD * SEQ;

  const v16h aq0 = frag_ld_f16(qtile,      HEADD, lane);   // d = 0..31
  const v16h aq1 = frag_ld_f16(qtile + 32, HEADD, lane);   // d = 32..63

  v8f o0 = {}, o1 = {}, o2 = {}, o3 = {};
  float mrun[8], lrun[8];
  #pragma unroll
  for (int v = 0; v < 8; ++v) { mrun[v] = -1e30f; lrun[v] = 0.0f; }

  _Float16* pl = lds_p[wslot];
  const int kh = (lane >> 4) & 1;
  const int cl = lane & 15;

  for (int kc = 0; kc < SEQ; kc += 32) {
    const _Float16* krow = kbase + (size_t)kc * HEADD;
    v16h b0a = frag_ld_f16(krow,                   HEADD, lane);
    v16h b0b = frag_ld_f16(krow + 32,              HEADD, lane);
    v16h b1a = frag_ld_f16(krow + 16 * HEADD,      HEADD, lane);
    v16h b1b = frag_ld_f16(krow + 16 * HEADD + 32, HEADD, lane);
    v8f s0 = {}, s1 = {};
    s0 = wmma_f16(aq0, b0a, s0);
    s0 = wmma_f16(aq1, b0b, s0);     // keys kc..kc+15
    s1 = wmma_f16(aq0, b1a, s1);
    s1 = wmma_f16(aq1, b1b, s1);     // keys kc+16..kc+31

    // Online softmax. Row stats per C-fragment VGPR (row = v + 8*kh);
    // columns striped over the 16-lane half -> xor-reduce masks 1..8.
    #pragma unroll
    for (int v = 0; v < 8; ++v) {
      const float x0 = s0[v] * 0.125f;     // 1/sqrt(64)
      const float x1 = s1[v] * 0.125f;
      float rmax = fmaxf(x0, x1);
      #pragma unroll
      for (int off = 1; off < 16; off <<= 1)
        rmax = fmaxf(rmax, __shfl_xor(rmax, off, 32));
      const float mnew  = fmaxf(mrun[v], rmax);
      const float alpha = __expf(mrun[v] - mnew);
      const float p0 = __expf(x0 - mnew);
      const float p1 = __expf(x1 - mnew);
      float rsum = p0 + p1;
      #pragma unroll
      for (int off = 1; off < 16; off <<= 1)
        rsum += __shfl_xor(rsum, off, 32);
      lrun[v] = lrun[v] * alpha + rsum;
      mrun[v] = mnew;
      o0[v] *= alpha; o1[v] *= alpha; o2[v] *= alpha; o3[v] *= alpha;
      const int rl = v + 8 * kh;           // C-layout -> LDS (row-major 16x32)
      pl[rl * 32 + cl]      = (_Float16)p0;
      pl[rl * 32 + 16 + cl] = (_Float16)p1;
    }
    asm volatile("s_wait_dscnt 0x0" ::: "memory");  // LDS transpose complete

    v16h ap  = frag_ld_f16(pl, 32, lane);                       // P as A (K=32)
    v16h bv0 = frag_ld_f16(vtbase +            kc, SEQ, lane);  // Vt rows d 0..15
    v16h bv1 = frag_ld_f16(vtbase + 16 * SEQ + kc, SEQ, lane);
    v16h bv2 = frag_ld_f16(vtbase + 32 * SEQ + kc, SEQ, lane);
    v16h bv3 = frag_ld_f16(vtbase + 48 * SEQ + kc, SEQ, lane);
    o0 = wmma_f16(ap, bv0, o0);
    o1 = wmma_f16(ap, bv1, o1);
    o2 = wmma_f16(ap, bv2, o2);
    o3 = wmma_f16(ap, bv3, o3);
  }

  const int h = bh % NHEADS;
  const int b = bh / NHEADS;
  #pragma unroll
  for (int v = 0; v < 8; ++v) {
    const float inv = 1.0f / lrun[v];
    const int r = b * SEQ + n0 + v + 8 * kh;
    _Float16* orow = attnbuf + (size_t)r * HIDDEN + h * HEADD;
    orow[ 0 + cl] = (_Float16)(o0[v] * inv);
    orow[16 + cl] = (_Float16)(o1[v] * inv);
    orow[32 + cl] = (_Float16)(o2[v] * inv);
    orow[48 + cl] = (_Float16)(o3[v] * inv);
  }
}

// ------------- Kernel 3: output projection, 32x64 tile per wave -------------
__global__ __launch_bounds__(256)
void proj_kernel(const _Float16* __restrict__ attnbuf, const _Float16* __restrict__ wo16,
                 float* __restrict__ out) {
  const int lane   = threadIdx.x & 31;
  const int wid    = blockIdx.x * 8 + (threadIdx.x >> 5);
  const int tile_m = wid % (ROWS / 32);   // 0..255
  const int tn64   = wid / (ROWS / 32);   // 0..11
  const _Float16* arow = attnbuf + (size_t)tile_m * 32 * HIDDEN;
  const _Float16* brow = wo16    + (size_t)tn64 * 64 * HIDDEN;

  v8f acc[2][4] = {};
  for (int k0 = 0; k0 < HIDDEN; k0 += 32) {
    __builtin_prefetch(arow + k0 + 64, 0, 1);
    v16h a0 = frag_ld_f16(arow + k0,               HIDDEN, lane);
    v16h a1 = frag_ld_f16(arow + 16 * HIDDEN + k0, HIDDEN, lane);
    v16h b0 = frag_ld_f16(brow + k0,               HIDDEN, lane);
    v16h b1 = frag_ld_f16(brow + 16 * HIDDEN + k0, HIDDEN, lane);
    v16h b2 = frag_ld_f16(brow + 32 * HIDDEN + k0, HIDDEN, lane);
    v16h b3 = frag_ld_f16(brow + 48 * HIDDEN + k0, HIDDEN, lane);
    acc[0][0] = wmma_f16(a0, b0, acc[0][0]);
    acc[0][1] = wmma_f16(a0, b1, acc[0][1]);
    acc[0][2] = wmma_f16(a0, b2, acc[0][2]);
    acc[0][3] = wmma_f16(a0, b3, acc[0][3]);
    acc[1][0] = wmma_f16(a1, b0, acc[1][0]);
    acc[1][1] = wmma_f16(a1, b1, acc[1][1]);
    acc[1][2] = wmma_f16(a1, b2, acc[1][2]);
    acc[1][3] = wmma_f16(a1, b3, acc[1][3]);
  }
  const int kh   = (lane >> 4) & 1;
  const int cl   = lane & 15;
  const int col0 = tn64 * 64 + cl;
  #pragma unroll
  for (int mi = 0; mi < 2; ++mi) {
    #pragma unroll
    for (int v = 0; v < 8; ++v) {
      const int r = tile_m * 32 + mi * 16 + v + 8 * kh;
      float* orow = out + (size_t)r * HIDDEN + col0;
      orow[ 0] = acc[mi][0][v];
      orow[16] = acc[mi][1][v];
      orow[32] = acc[mi][2][v];
      orow[48] = acc[mi][3][v];
    }
  }
}

extern "C" void kernel_launch(void* const* d_in, const int* in_sizes, int n_in,
                              void* d_out, int out_size, void* d_ws, size_t ws_size,
                              hipStream_t stream) {
  const float* x     = (const float*)d_in[0];
  const float* w_qkv = (const float*)d_in[1];
  const float* w_out = (const float*)d_in[2];
  float* out = (float*)d_out;

  const size_t per     = (size_t)BATCH * NHEADS * SEQ * HEADD;  // 6,291,456
  const size_t n_x     = (size_t)ROWS * HIDDEN;                 // 6,291,456
  const size_t n_wqkv  = (size_t)QKV_OUT * HIDDEN;              // 1,769,472
  const size_t n_wout  = (size_t)HIDDEN * HIDDEN;               //   589,824

  _Float16* qbuf    = (_Float16*)d_ws;
  _Float16* kbuf    = qbuf + per;
  _Float16* vtbuf   = kbuf + per;
  _Float16* attnbuf = vtbuf + per;
  _Float16* x16     = attnbuf + n_x;
  _Float16* wqkv16  = x16 + n_x;
  _Float16* wout16  = wqkv16 + n_wqkv;

  // One-time fp32 -> f16 conversion (O(elements), amortized over all tiles)
  cvt_kernel<<<(int)(n_x    / 4 / 256), 256, 0, stream>>>(x,     x16,    (int)(n_x / 4));
  cvt_kernel<<<(int)(n_wqkv / 4 / 256), 256, 0, stream>>>(w_qkv, wqkv16, (int)(n_wqkv / 4));
  cvt_kernel<<<(int)(n_wout / 4 / 256), 256, 0, stream>>>(w_out, wout16, (int)(n_wout / 4));

  // 256 m-tiles * 36 n64-tiles / 8 waves per block
  qkv_kernel<<<(ROWS / 32) * (QKV_OUT / 64) / 8, 256, 0, stream>>>(x16, wqkv16, qbuf, kbuf, vtbuf);
  // 24 (b*h) * 256 row-tiles / 8 waves per block
  attn_kernel<<<BATCH * NHEADS * (SEQ / 16) / 8, 256, 0, stream>>>(qbuf, kbuf, vtbuf, attnbuf);
  // 256 m-tiles * 12 n64-tiles / 8 waves per block
  proj_kernel<<<(ROWS / 32) * (HIDDEN / 64) / 8, 256, 0, stream>>>(attnbuf, wout16, out);
}
